// Decoder_89550068122002
// MI455X (gfx1250) — matrix operand order repaired and checked by reading, not compiled, not run on hardware
//
#include <hip/hip_runtime.h>
#include <hip/hip_bf16.h>

// ---------------------------------------------------------------------------
// Types for CDNA5 WMMA (wave32): v_wmma_f32_16x16x32_bf16
// ---------------------------------------------------------------------------
typedef __attribute__((ext_vector_type(16))) __bf16 v16bf;
typedef __attribute__((ext_vector_type(8)))  __bf16 v8bf;
typedef __attribute__((ext_vector_type(4)))  __bf16 v4bf;
typedef __attribute__((ext_vector_type(8)))  float  v8f;
typedef __attribute__((ext_vector_type(4)))  float  v4f;

__device__ __forceinline__ float sigmoidf_(float x) { return 1.0f / (1.0f + __expf(-x)); }

__device__ __forceinline__ v4bf cvt4(v4f v) {
    v4bf p;
    p[0] = (__bf16)v[0]; p[1] = (__bf16)v[1];
    p[2] = (__bf16)v[2]; p[3] = (__bf16)v[3];
    return p;
}

// ---------------------------------------------------------------------------
// Elementwise helpers
// ---------------------------------------------------------------------------
__global__ void k_f32_to_bf16(const float* __restrict__ src, __bf16* __restrict__ dst, long n) {
    long i = (long)blockIdx.x * blockDim.x + threadIdx.x;
    if (i < n) dst[i] = (__bf16)src[i];
}

__global__ void k_copy_f32(const float* __restrict__ src, float* __restrict__ dst, long n) {
    long i = (long)blockIdx.x * blockDim.x + threadIdx.x;
    if (i < n) dst[i] = src[i];
}

// dst[b][h][s] = bf16(hs[b][s][h])   (hs: [B,S,H] -> hsT: [B,H,S])
__global__ void k_hs_transpose_bf16(const float* __restrict__ hs, __bf16* __restrict__ dst,
                                    int S, int H, long n) {
    long i = (long)blockIdx.x * blockDim.x + threadIdx.x;
    if (i >= n) return;
    int  s = (int)(i % S);
    long r = i / S;
    int  h = (int)(r % H);
    int  b = (int)(r / H);
    dst[i] = (__bf16)hs[((long)b * S + s) * H + h];
}

// cat[row][0:H] = out1[row], cat[row][H:2H] = ctx[row]
__global__ void k_concat(const float* __restrict__ out1, const float* __restrict__ ctx,
                         float* __restrict__ cat, int H, long n) {
    long i = (long)blockIdx.x * blockDim.x + threadIdx.x;
    if (i >= n) return;
    long row = i / (2 * H);
    int  col = (int)(i % (2 * H));
    cat[i] = (col < H) ? out1[row * H + col] : ctx[row * H + (col - H)];
}

// ---------------------------------------------------------------------------
// Generic WMMA GEMM:  C[M,N] = act( A_f32[M,K] @ W_bf16[N,K]^T + bias[N] )
// REQUIRES: M % 64 == 0, N % 64 == 0, K % 32 == 0 (true for every call here).
// Batched via blockIdx.z with element strides sA/sW/sC. Block tile 64x64,
// 256 threads = 8 wave32 waves; each wave computes a 16x32 strip.
// ---------------------------------------------------------------------------
#define BM 64
#define BN 64
#define BKK 32

__global__ __launch_bounds__(256) void k_gemm_bf16wmma(
    const float* __restrict__ A, int lda, long sA,
    const __bf16* __restrict__ W, long sW,
    const float* __restrict__ bias,
    float* __restrict__ C, int ldc, long sC,
    int M, int N, int K, int act)
{
    __shared__ __align__(32) __bf16 sAt[BM * BKK];
    __shared__ __align__(32) __bf16 sBt[BN * BKK];

    const int bz = blockIdx.z;
    const float*  Ab = A + (long)bz * sA;
    const __bf16* Wb = W + (long)bz * sW;
    float*        Cb = C + (long)bz * sC;

    const int n0 = blockIdx.x * BN;
    const int m0 = blockIdx.y * BM;
    const int tid  = threadIdx.x;
    const int w    = tid >> 5;
    const int lane = tid & 31;
    const int wr = w >> 1;           // 0..3  (M sub-tile)
    const int wc = w & 1;            // 0..1  (N half, 32 cols)
    const int lr  = lane & 15;
    const int lhi = lane >> 4;

    // Staging coordinates (hoisted)
    const int ar  = tid >> 3;              // A row for j=0 (0..31)
    const int ac  = (tid & 7) * 4;         // A col chunk
    const int bn  = tid >> 2;              // B row (0..63)
    const int bkk = (tid & 3) * 8;         // B col chunk

    v8f acc0 = {};
    v8f acc1 = {};

    for (int k0 = 0; k0 < K; k0 += BKK) {
        __syncthreads();
        // Stage A tile 64x32: float4 loads -> 4x bf16 -> 8B LDS stores
        #pragma unroll
        for (int j = 0; j < 2; ++j) {
            int r = ar + 32 * j;
            v4f v = *(const v4f*)(Ab + (long)(m0 + r) * lda + k0 + ac);
            *((v4bf*)&sAt[r * BKK + ac]) = cvt4(v);
        }
        // Stage B tile 64x32 bf16, 16B vector copies
        *((v8bf*)&sBt[bn * BKK + bkk]) =
            *(const v8bf*)(Wb + (long)(n0 + bn) * K + k0 + bkk);
        // Prefetch next weight tile (global_prefetch_b8)
        if (k0 + BKK < K)
            __builtin_prefetch((const void*)(Wb + (long)(n0 + bn) * K + k0 + BKK + bkk), 0, 3);
        __syncthreads();

        // A fragment, contiguous K order: lanes 0-15 -> K{0..15}, 16-31 -> K{16..31}
        const int arow = 16 * wr + lr;
        const int akb  = lhi ? 16 : 0;
        v8bf alo = *(const v8bf*)&sAt[arow * BKK + akb];
        v8bf ahi = *(const v8bf*)&sAt[arow * BKK + akb + 8];
        v16bf afrag;
        #pragma unroll
        for (int i = 0; i < 8; ++i) { afrag[i] = alo[i]; afrag[i + 8] = ahi[i]; }

        const int bkb = lhi ? 16 : 0;
        {
            int bcol = 32 * wc + lr;
            v16bf bfrag = *(const v16bf*)&sBt[bcol * BKK + bkb];
            acc0 = __builtin_amdgcn_wmma_f32_16x16x32_bf16(false, afrag, false, bfrag,
                                                           (short)0, acc0, false, false);
        }
        {
            int bcol = 32 * wc + 16 + lr;
            v16bf bfrag = *(const v16bf*)&sBt[bcol * BKK + bkb];
            acc1 = __builtin_amdgcn_wmma_f32_16x16x32_bf16(false, afrag, false, bfrag,
                                                           (short)0, acc1, false, false);
        }
    }

    // Epilogue: C layout VGPR r -> M=r (lanes 0-15), M=r+8 (lanes 16-31); N = lane%16
    #pragma unroll
    for (int r = 0; r < 8; ++r) {
        int m = m0 + 16 * wr + r + 8 * lhi;
        #pragma unroll
        for (int j = 0; j < 2; ++j) {
            int n = n0 + 32 * wc + 16 * j + lr;
            float v = (j == 0) ? acc0[r] : acc1[r];
            if (bias) v += bias[n];
            if (act == 1) v = tanhf(v);
            Cb[(long)m * ldc + n] = v;
        }
    }
}

// ---------------------------------------------------------------------------
// Fused GRU step: gh = h_prev @ w_hh^T + b_hh for all 3 gates of 64 h-columns,
// then gate math. Grid = H/64 blocks, 256 threads (8 waves: 2 M-rows x 4 N-cols).
// B (batch) = 32 rows. xp holds precomputed input projections [B*T, 3H].
// ---------------------------------------------------------------------------
__global__ __launch_bounds__(256) void k_gru_step(
    const float* __restrict__ xp,
    const float* __restrict__ hprev,     // [32, H]
    const __bf16* __restrict__ whh,      // [3H, H] bf16
    const float* __restrict__ bhh,       // [3H]
    float* __restrict__ hnext,           // [32, H]
    float* __restrict__ out,             // [B*T, H]
    int T, int t, int H)
{
    __shared__ __align__(32) __bf16 sA[32 * 32];
    __shared__ __align__(32) __bf16 sB[3][64 * 32];

    const int c0 = blockIdx.x * 64;
    const int tid  = threadIdx.x;
    const int w    = tid >> 5;
    const int lane = tid & 31;
    const int wr = w >> 2;           // 0..1 (batch rows 16*wr..)
    const int wc = w & 3;            // 0..3 (16-col tile)
    const int lr  = lane & 15;
    const int lhi = lane >> 4;

    const int ar  = tid >> 3;              // 0..31
    const int ac  = (tid & 7) * 4;
    const int bn  = tid >> 2;
    const int bkk = (tid & 3) * 8;

    v8f accR = {}, accZ = {}, accN = {};

    for (int k0 = 0; k0 < H; k0 += 32) {
        __syncthreads();
        // A: h_prev tile 32x32 (float4 -> bf16x4)
        {
            v4f v = *(const v4f*)(hprev + (long)ar * H + k0 + ac);
            *((v4bf*)&sA[ar * 32 + ac]) = cvt4(v);
        }
        // B: 3 gate weight tiles 64x32 bf16
        #pragma unroll
        for (int g = 0; g < 3; ++g) {
            *((v8bf*)&sB[g][bn * 32 + bkk]) =
                *(const v8bf*)(whh + (long)(g * H + c0 + bn) * H + k0 + bkk);
        }
        if (k0 + 32 < H) {
            __builtin_prefetch((const void*)(whh + (long)(c0 + bn) * H + k0 + 32 + bkk), 0, 3);
        }
        __syncthreads();

        const int arow = 16 * wr + lr;
        const int akb  = lhi ? 16 : 0;
        v8bf alo = *(const v8bf*)&sA[arow * 32 + akb];
        v8bf ahi = *(const v8bf*)&sA[arow * 32 + akb + 8];
        v16bf af;
        #pragma unroll
        for (int i = 0; i < 8; ++i) { af[i] = alo[i]; af[i + 8] = ahi[i]; }

        const int bkb  = lhi ? 16 : 0;
        const int bcol = 16 * wc + lr;
        v16bf bR = *(const v16bf*)&sB[0][bcol * 32 + bkb];
        v16bf bZ = *(const v16bf*)&sB[1][bcol * 32 + bkb];
        v16bf bN = *(const v16bf*)&sB[2][bcol * 32 + bkb];
        accR = __builtin_amdgcn_wmma_f32_16x16x32_bf16(false, af, false, bR, (short)0, accR, false, false);
        accZ = __builtin_amdgcn_wmma_f32_16x16x32_bf16(false, af, false, bZ, (short)0, accZ, false, false);
        accN = __builtin_amdgcn_wmma_f32_16x16x32_bf16(false, af, false, bN, (short)0, accN, false, false);
    }

    // Gate math + state update
    #pragma unroll
    for (int r = 0; r < 8; ++r) {
        int bRow = 16 * wr + r + 8 * lhi;          // batch index 0..31
        int col  = c0 + 16 * wc + lr;              // h column
        long xrow = ((long)bRow * T + t) * (3L * H);
        float xr = xp[xrow + col];
        float xz = xp[xrow + H + col];
        float xn = xp[xrow + 2L * H + col];
        float gr = accR[r] + bhh[col];
        float gz = accZ[r] + bhh[H + col];
        float gn = accN[r] + bhh[2 * H + col];
        float rg = sigmoidf_(xr + gr);
        float zg = sigmoidf_(xz + gz);
        float ng = tanhf(xn + rg * gn);
        float hp = hprev[(long)bRow * H + col];
        float hv = (1.0f - zg) * ng + zg * hp;
        hnext[(long)bRow * H + col] = hv;
        out[((long)bRow * T + t) * H + col] = hv;
    }
}

// ---------------------------------------------------------------------------
// Masked softmax over encoder dim (S=128). One block per (b,t), 128 threads.
// In-place on s[B,S,T]; also writes transposed attn_t[B,T,S] for context GEMM.
// ---------------------------------------------------------------------------
__global__ __launch_bounds__(128) void k_softmax_attn(
    float* __restrict__ s, float* __restrict__ attn_t,
    const int* __restrict__ mask, int S, int T)
{
    __shared__ float red[128];
    int bt = blockIdx.x;
    int b = bt / T, t = bt % T;
    int si = threadIdx.x;
    long idx = ((long)b * S + si) * T + t;
    int mlen = mask[b];
    float v = (si < mlen) ? s[idx] : -__builtin_inff();

    red[si] = v; __syncthreads();
    for (int o = 64; o > 0; o >>= 1) {
        if (si < o) red[si] = fmaxf(red[si], red[si + o]);
        __syncthreads();
    }
    float mx = red[0]; __syncthreads();

    float e = (si < mlen) ? __expf(v - mx) : 0.0f;
    red[si] = e; __syncthreads();
    for (int o = 64; o > 0; o >>= 1) {
        if (si < o) red[si] += red[si + o];
        __syncthreads();
    }
    float a = e / red[0];
    s[idx] = a;
    attn_t[((long)b * T + t) * S + si] = a;
}

// ---------------------------------------------------------------------------
// Host driver
// ---------------------------------------------------------------------------
extern "C" void kernel_launch(void* const* d_in, const int* in_sizes, int n_in,
                              void* d_out, int out_size, void* d_ws, size_t ws_size,
                              hipStream_t stream)
{
    (void)in_sizes; (void)n_in; (void)out_size; (void)ws_size;
    const int B = 32, T = 64, S = 128, E = 512, H = 1024, V = 32000;
    const long BT = (long)B * T;

    const float* emb     = (const float*)d_in[0];
    const float* hs      = (const float*)d_in[1];
    const float* h_in    = (const float*)d_in[2];
    const int*   mask    = (const int*)  d_in[3];
    const float* w_ih0   = (const float*)d_in[4];
    const float* w_hh0   = (const float*)d_in[5];
    const float* b_ih0   = (const float*)d_in[6];
    const float* b_hh0   = (const float*)d_in[7];
    const float* w_ih1   = (const float*)d_in[8];
    const float* w_hh1   = (const float*)d_in[9];
    const float* b_ih1   = (const float*)d_in[10];
    const float* b_hh1   = (const float*)d_in[11];
    const float* cat_w   = (const float*)d_in[12];
    const float* cat_b   = (const float*)d_in[13];
    const float* out_w   = (const float*)d_in[14];
    const float* out_b   = (const float*)d_in[15];

    float* out = (float*)d_out;
    float* out_logits = out;                                    // [B,T,V]
    float* out_hfinal = out + (long)B * T * V;                  // [2,B,H]
    float* out_attn   = out_hfinal + 2L * B * H;                // [B,S,T]

    // Workspace layout
    char* ws = (char*)d_ws;
    size_t off = 0;
    auto alloc = [&](size_t bytes) -> char* {
        char* p = ws + off;
        off = (off + bytes + 255) & ~(size_t)255;
        return p;
    };
    __bf16* wih0_b = (__bf16*)alloc((size_t)3 * H * E * 2);
    __bf16* whh0_b = (__bf16*)alloc((size_t)3 * H * H * 2);
    __bf16* wih1_b = (__bf16*)alloc((size_t)3 * H * H * 2);
    __bf16* whh1_b = (__bf16*)alloc((size_t)3 * H * H * 2);
    __bf16* catw_b = (__bf16*)alloc((size_t)H * 2 * H * 2);
    __bf16* outw_b = (__bf16*)alloc((size_t)V * H * 2);
    float*  xp     = (float*) alloc((size_t)BT * 3 * H * 4);    // reused: layer0 xp, layer1 xp, then cat
    float*  out0   = (float*) alloc((size_t)BT * H * 4);
    float*  out1   = (float*) alloc((size_t)BT * H * 4);
    __bf16* out1_b = (__bf16*)alloc((size_t)BT * H * 2);
    __bf16* hsT_b  = (__bf16*)alloc((size_t)B * H * S * 2);
    float*  attn_t = (float*) alloc((size_t)B * T * S * 4);
    float*  ctx    = (float*) alloc((size_t)BT * H * 4);
    float*  obuf   = (float*) alloc((size_t)BT * H * 4);
    float*  h0a    = (float*) alloc((size_t)B * H * 4);
    float*  h0b    = (float*) alloc((size_t)B * H * 4);
    float*  h1a    = (float*) alloc((size_t)B * H * 4);
    float*  h1b    = (float*) alloc((size_t)B * H * 4);

    auto grid1d = [](long n) { return dim3((unsigned)((n + 255) / 256)); };

    // 1) Convert weights to bf16 once (cheap; dominated by out_w 32M elems)
    k_f32_to_bf16<<<grid1d(3L*H*E),  256, 0, stream>>>(w_ih0, wih0_b, 3L*H*E);
    k_f32_to_bf16<<<grid1d(3L*H*H),  256, 0, stream>>>(w_hh0, whh0_b, 3L*H*H);
    k_f32_to_bf16<<<grid1d(3L*H*H),  256, 0, stream>>>(w_ih1, wih1_b, 3L*H*H);
    k_f32_to_bf16<<<grid1d(3L*H*H),  256, 0, stream>>>(w_hh1, whh1_b, 3L*H*H);
    k_f32_to_bf16<<<grid1d((long)H*2*H), 256, 0, stream>>>(cat_w, catw_b, (long)H*2*H);
    k_f32_to_bf16<<<grid1d((long)V*H),   256, 0, stream>>>(out_w, outw_b, (long)V*H);

    // 2) Layer-0 input projections: xp = emb[2048,512] @ w_ih0^T + b_ih0
    k_gemm_bf16wmma<<<dim3(3*H/64, BT/64, 1), 256, 0, stream>>>(
        emb, E, 0, wih0_b, 0, b_ih0, xp, 3*H, 0, (int)BT, 3*H, E, 0);

    // 3) GRU layer 0 (sequential over T), double-buffered hidden state
    k_copy_f32<<<grid1d((long)B*H), 256, 0, stream>>>(h_in,              h0a, (long)B*H);
    k_copy_f32<<<grid1d((long)B*H), 256, 0, stream>>>(h_in + (long)B*H,  h1a, (long)B*H);
    {
        float* hc = h0a; float* hn = h0b;
        for (int t = 0; t < T; ++t) {
            k_gru_step<<<dim3(H/64), 256, 0, stream>>>(xp, hc, whh0_b, b_hh0, hn, out0, T, t, H);
            float* tmp = hc; hc = hn; hn = tmp;
        }
        k_copy_f32<<<grid1d((long)B*H), 256, 0, stream>>>(hc, out_hfinal, (long)B*H);
    }

    // 4) Layer-1 input projections: xp = out0[2048,1024] @ w_ih1^T + b_ih1
    k_gemm_bf16wmma<<<dim3(3*H/64, BT/64, 1), 256, 0, stream>>>(
        out0, H, 0, wih1_b, 0, b_ih1, xp, 3*H, 0, (int)BT, 3*H, H, 0);

    // 5) GRU layer 1
    {
        float* hc = h1a; float* hn = h1b;
        for (int t = 0; t < T; ++t) {
            k_gru_step<<<dim3(H/64), 256, 0, stream>>>(xp, hc, whh1_b, b_hh1, hn, out1, T, t, H);
            float* tmp = hc; hc = hn; hn = tmp;
        }
        k_copy_f32<<<grid1d((long)B*H), 256, 0, stream>>>(hc, out_hfinal + (long)B*H, (long)B*H);
    }

    // 6) bf16 copies for attention GEMMs
    k_f32_to_bf16<<<grid1d(BT*(long)H), 256, 0, stream>>>(out1, out1_b, BT*(long)H);
    k_hs_transpose_bf16<<<grid1d((long)B*H*S), 256, 0, stream>>>(hs, hsT_b, S, H, (long)B*H*S);

    // 7) Attention scores (batched): s[b] = hs[b][S,H] @ out1[b][T,H]^T -> [S,T]
    k_gemm_bf16wmma<<<dim3(T/64, S/64, B), 256, 0, stream>>>(
        hs, H, (long)S*H, out1_b, (long)T*H, nullptr,
        out_attn, T, (long)S*T, S, T, H, 0);

    // 8) Masked softmax over encoder dim (in-place) + transposed copy
    k_softmax_attn<<<dim3((unsigned)(B*T)), 128, 0, stream>>>(out_attn, attn_t, mask, S, T);

    // 9) Context (batched): ctx[b] = attn_t[b][T,S] @ hsT[b][H,S]^T -> [T,H]
    k_gemm_bf16wmma<<<dim3(H/64, T/64, B), 256, 0, stream>>>(
        attn_t, S, (long)T*S, hsT_b, (long)H*S, nullptr,
        ctx, H, (long)T*H, T, H, S, 0);

    // 10) cat = [out1 | ctx]  (reuse xp buffer: [2048, 2048])
    float* cat = xp;
    k_concat<<<grid1d(BT*2L*H), 256, 0, stream>>>(out1, ctx, cat, H, BT*2L*H);

    // 11) o = tanh(cat @ concat_w^T + concat_b)
    k_gemm_bf16wmma<<<dim3(H/64, BT/64, 1), 256, 0, stream>>>(
        cat, 2*H, 0, catw_b, 0, cat_b, obuf, H, 0, (int)BT, H, 2*H, 1);

    // 12) logits = o @ out_w^T + out_b   (M=2048, N=32000, K=1024 — dominant GEMM)
    k_gemm_bf16wmma<<<dim3(V/64, BT/64, 1), 256, 0, stream>>>(
        obuf, H, 0, outw_b, 0, out_b, out_logits, V, 0, (int)BT, V, H, 0);
}
